// GPT2Attention_27496380629258
// MI455X (gfx1250) — matrix operand order repaired and checked by reading, not compiled, not run on hardware
//
#include <hip/hip_runtime.h>
#include <hip/hip_bf16.h>

// ---------------------------------------------------------------------------
// Problem constants (fixed by the reference)
// ---------------------------------------------------------------------------
#define BATCH   4
#define SEQ     2048
#define CDIM    1024
#define NHEAD   16
#define NKV     4
#define HDIM    64
#define NTOK    (BATCH * SEQ)            // 8192
#define QKVW    (CDIM + 2 * NKV * HDIM)  // 1536

typedef __bf16 bf16_t;
typedef __attribute__((ext_vector_type(16))) __bf16 v16bf;
typedef __attribute__((ext_vector_type(8)))  float  v8f;
typedef int v4i __attribute__((ext_vector_type(4)));
typedef __attribute__((address_space(1))) v4i gv4i;   // global int4 vector
typedef __attribute__((address_space(3))) v4i lv4i;   // LDS int4 vector

union Frag16 {                 // one WMMA 16-bit A/B operand (8 VGPRs)
    v16bf          v;
    unsigned short u[16];
    unsigned int   w32[8];
    uint4          q[2];
};
union F4 { float4 v; float f[4]; };

__device__ __forceinline__ unsigned short bf16_bits(float f) {
    union { __bf16 h; unsigned short u; } t;
    t.h = (__bf16)f;                       // hardware f32->bf16 convert (RNE)
    return t.u;
}
__device__ __forceinline__ unsigned int pack_bf16x2(float a, float b) {
    union { __bf16 h[2]; unsigned int u; } t;
    t.h[0] = (__bf16)a; t.h[1] = (__bf16)b;
    return t.u;
}

// A-matrix K index pattern for 16-bit 16x32 fragments (wave32):
// pair p (VGPR p), lane-half h: Kbase = (p<4 ? 2p : 16+2(p-4)) + 8h
__device__ __forceinline__ int a_kbase(int p, int half) {
    return ((p < 4) ? (2 * p) : (16 + 2 * (p - 4))) + 8 * half;
}

// ---------------------------------------------------------------------------
// CDNA5 async global->LDS copy (ASYNCcnt path), with safe fallback.
// ---------------------------------------------------------------------------
__device__ __forceinline__ void async_copy_b128(const void* g, void* lds) {
#if __has_builtin(__builtin_amdgcn_global_load_async_to_lds_b128)
    __builtin_amdgcn_global_load_async_to_lds_b128(
        (gv4i*)(size_t)g,
        (lv4i*)(unsigned int)(size_t)lds,
        0, 0);
#else
    *(uint4*)lds = *(const uint4*)g;       // synchronous fallback
#endif
}
__device__ __forceinline__ void async_wait0() {
#if __has_builtin(__builtin_amdgcn_s_wait_asynccnt)
    __builtin_amdgcn_s_wait_asynccnt(0);
#elif __has_builtin(__builtin_amdgcn_global_load_async_to_lds_b128)
    asm volatile("s_wait_asynccnt 0x0" ::: "memory");
#endif
}

// ---------------------------------------------------------------------------
// Kernel 1/4: generic  C[M,N] = A[M,K] * B[N,K]^T  (fp32 in/out, bf16 WMMA).
// Block tile 128x64, 8 waves, wave tile 32x32. All launch shapes divide the
// tiles exactly -> no bounds checks.
// ---------------------------------------------------------------------------
#define GM_BM 128
#define GM_BN 64
#define GM_BK 32
#define GM_LP 36          // LDS row stride (ushorts): 72B, 8B-aligned rows

__global__ __launch_bounds__(256)
void gemm_bf16_wmma(const float* __restrict__ A, const float* __restrict__ Bw,
                    float* __restrict__ C, int K, int ldc, int coff) {
    __shared__ unsigned short sA[GM_BM * GM_LP];
    __shared__ unsigned short sB[GM_BN * GM_LP];

    const int tid   = threadIdx.x;
    const int lane  = tid & 31;
    const int wave  = tid >> 5;
    const int wm    = wave >> 1;          // 0..3  (M sub-tile)
    const int wn    = wave & 1;           // 0..1  (N sub-tile)
    const int col16 = lane & 15;
    const int half  = lane >> 4;
    const int bm0   = blockIdx.y * GM_BM;
    const int bn0   = blockIdx.x * GM_BN;

    // staging coordinates (float4 granularity)
    const int ra = tid >> 3, ca = tid & 7;        // A: 4 iters of 32 rows
    const float4* Ag = (const float4*)(A + (size_t)(bm0 + ra) * K) + ca;
    const float4* Bg = (const float4*)(Bw + (size_t)(bn0 + ra) * K) + ca;
    const size_t rowstep4 = (size_t)K >> 2;       // float4s per row

    v8f acc[2][2];
    const v8f vzero = {0.f, 0.f, 0.f, 0.f, 0.f, 0.f, 0.f, 0.f};
    acc[0][0] = vzero; acc[0][1] = vzero; acc[1][0] = vzero; acc[1][1] = vzero;

    for (int k0 = 0; k0 < K; k0 += GM_BK) {
        const size_t k4 = (size_t)(k0 >> 2);
        // ---- stage A tile (128 x 32) as bf16 ----
        #pragma unroll
        for (int it = 0; it < 4; ++it) {
            F4 a; a.v = Ag[(size_t)(it * 32) * rowstep4 + k4];
            uint2 p; p.x = pack_bf16x2(a.f[0], a.f[1]);
                     p.y = pack_bf16x2(a.f[2], a.f[3]);
            *(uint2*)&sA[(it * 32 + ra) * GM_LP + ca * 4] = p;
        }
        // ---- stage B tile (64 x 32) ----
        #pragma unroll
        for (int it = 0; it < 2; ++it) {
            F4 a; a.v = Bg[(size_t)(it * 32) * rowstep4 + k4];
            uint2 p; p.x = pack_bf16x2(a.f[0], a.f[1]);
                     p.y = pack_bf16x2(a.f[2], a.f[3]);
            *(uint2*)&sB[(it * 32 + ra) * GM_LP + ca * 4] = p;
        }
        if (k0 + GM_BK < K) {                     // prefetch next K tile
            __builtin_prefetch(&Ag[k4 + 8], 0, 1);
            __builtin_prefetch(&Bg[k4 + 8], 0, 1);
        }
        __syncthreads();

        Frag16 af[2], bfr[2];
        #pragma unroll
        for (int mi = 0; mi < 2; ++mi) {
            const unsigned short* rp = &sA[(wm * 32 + mi * 16 + col16) * GM_LP];
            #pragma unroll
            for (int p = 0; p < 8; ++p)
                af[mi].w32[p] = *(const unsigned int*)(rp + a_kbase(p, half));
        }
        #pragma unroll
        for (int ni = 0; ni < 2; ++ni) {
            const unsigned short* rp =
                &sB[(wn * 32 + ni * 16 + col16) * GM_LP];
            #pragma unroll
            for (int p = 0; p < 8; ++p)
                bfr[ni].w32[p] = *(const unsigned int*)(rp + a_kbase(p, half));
        }
        #pragma unroll
        for (int mi = 0; mi < 2; ++mi)
            #pragma unroll
            for (int ni = 0; ni < 2; ++ni)
                acc[mi][ni] = __builtin_amdgcn_wmma_f32_16x16x32_bf16(
                    false, af[mi].v, false, bfr[ni].v,
                    (short)0, acc[mi][ni], false, false);
        __syncthreads();
    }

    // ---- store: one base pointer, row stepping, immediate column offsets ----
    float* rowp = C + (size_t)(bm0 + wm * 32 + 8 * half) * ldc
                    + (coff + bn0 + wn * 32 + col16);
    #pragma unroll
    for (int mi = 0; mi < 2; ++mi) {
        float* rp = rowp + (size_t)(mi * 16) * ldc;
        #pragma unroll
        for (int v = 0; v < 8; ++v) {
            rp[0]  = acc[mi][0][v];
            rp[16] = acc[mi][1][v];
            rp += ldc;
        }
    }
}

// ---------------------------------------------------------------------------
// Kernel 2/4: epilogue — gate(ve) add on V; rotary + RMS-norm * 1.15 on Q,K;
// emit bf16 tensors laid out [B,H,T,D] (q) and [B,KV,T,D] (k,v).
// ---------------------------------------------------------------------------
__global__ __launch_bounds__(256)
void qkv_epilogue(const float* __restrict__ qkv, const float* __restrict__ x,
                  const float* __restrict__ ve, const float* __restrict__ cosb,
                  const float* __restrict__ sinb,
                  const float* __restrict__ w_gate,
                  unsigned short* __restrict__ qb,
                  unsigned short* __restrict__ kb,
                  unsigned short* __restrict__ vb) {
    int j = blockIdx.x * blockDim.x + threadIdx.x;
    if (j >= NTOK * (NHEAD + 2 * NKV)) return;
    const int token = j / (NHEAD + 2 * NKV);
    const int u     = j - token * (NHEAD + 2 * NKV);
    const int b     = token / SEQ;
    const int t     = token - b * SEQ;

    if (u < NHEAD + NKV) {                       // q or k head: rotary + rms
        const int off = (u < NHEAD) ? (u * HDIM) : (CDIM + (u - NHEAD) * HDIM);
        const float4* s4  = (const float4*)(qkv + (size_t)token * QKVW + off);
        const float4* c4  = (const float4*)(cosb + t * (HDIM / 2));
        const float4* sn4 = (const float4*)(sinb + t * (HDIM / 2));
        float o[HDIM];
        float ss = 0.f;
        #pragma unroll
        for (int g = 0; g < 8; ++g) {
            F4 x1, x2, c, s;
            x1.v = s4[g]; x2.v = s4[g + 8]; c.v = c4[g]; s.v = sn4[g];
            #pragma unroll
            for (int e = 0; e < 4; ++e) {
                float o1 =  x1.f[e] * c.f[e] + x2.f[e] * s.f[e];
                float o2 = -x1.f[e] * s.f[e] + x2.f[e] * c.f[e];
                o[g * 4 + e] = o1; o[32 + g * 4 + e] = o2;
                ss += o1 * o1 + o2 * o2;
            }
        }
        float r = rsqrtf(ss * (1.0f / HDIM) + 1e-6f) * 1.15f;
        unsigned short* dst =
            (u < NHEAD)
                ? qb + (((size_t)b * NHEAD + u) * SEQ + t) * HDIM
                : kb + (((size_t)b * NKV + (u - NHEAD)) * SEQ + t) * HDIM;
        uint4* d4 = (uint4*)dst;
        #pragma unroll
        for (int g = 0; g < 8; ++g) {
            uint4 w;
            w.x = pack_bf16x2(o[g * 8 + 0] * r, o[g * 8 + 1] * r);
            w.y = pack_bf16x2(o[g * 8 + 2] * r, o[g * 8 + 3] * r);
            w.z = pack_bf16x2(o[g * 8 + 4] * r, o[g * 8 + 5] * r);
            w.w = pack_bf16x2(o[g * 8 + 6] * r, o[g * 8 + 7] * r);
            d4[g] = w;
        }
    } else {                                     // v head: gated ve add
        const int kv = u - NHEAD - NKV;
        float g = 0.f;
        const float4* xg = (const float4*)(x + (size_t)token * CDIM);
        #pragma unroll
        for (int c = 0; c < 3; ++c) {
            F4 xa, wa;
            xa.v = xg[c]; wa.v = ((const float4*)(w_gate + kv * 12))[c];
            #pragma unroll
            for (int e = 0; e < 4; ++e) g += xa.f[e] * wa.f[e];
        }
        g = 3.0f / (1.0f + __expf(-g));
        const float4* vs = (const float4*)(qkv + (size_t)token * QKVW + CDIM +
                                           NKV * HDIM + kv * HDIM);
        const float4* es = (const float4*)(ve + (size_t)token * (NKV * HDIM) +
                                           kv * HDIM);
        uint4* d4 = (uint4*)(vb + (((size_t)b * NKV + kv) * SEQ + t) * HDIM);
        #pragma unroll
        for (int q2 = 0; q2 < 4; ++q2) {         // 8 float4 in -> 4 uint4 out
            F4 a, b2, ea, eb;
            a.v  = vs[q2 * 2];     b2.v = vs[q2 * 2 + 1];
            ea.v = es[q2 * 2];     eb.v = es[q2 * 2 + 1];
            uint4 w;
            w.x = pack_bf16x2(a.f[0]  + g * ea.f[0], a.f[1]  + g * ea.f[1]);
            w.y = pack_bf16x2(a.f[2]  + g * ea.f[2], a.f[3]  + g * ea.f[3]);
            w.z = pack_bf16x2(b2.f[0] + g * eb.f[0], b2.f[1] + g * eb.f[1]);
            w.w = pack_bf16x2(b2.f[2] + g * eb.f[2], b2.f[3] + g * eb.f[3]);
            d4[q2] = w;
        }
    }
}

// ---------------------------------------------------------------------------
// Kernel 3/4: sliding-window flash attention, bf16 WMMA, online softmax.
// K and V tiles staged via async global->LDS (ASYNCcnt), P re-laid through
// per-wave LDS. Block = (q-tile 64) x head x batch; 4 waves.
// ---------------------------------------------------------------------------
#define AT_KS 72     // LDS K/V row stride (ushorts): 144B rows, 16B-aligned
#define AT_PS 68     // LDS P row stride (ushorts)

__global__ __launch_bounds__(128)
void attn_window_wmma(const unsigned short* __restrict__ qb,
                      const unsigned short* __restrict__ kb,
                      const unsigned short* __restrict__ vb,
                      float* __restrict__ y,
                      const int* __restrict__ winp) {
    __shared__ unsigned short sK[64 * AT_KS];          // K tile, [key][d]
    __shared__ unsigned short sV[64 * AT_KS];          // V tile, [key][d]
    __shared__ unsigned short sP[4][16 * AT_PS];       // per-wave P, [m][key]

    const int tid   = threadIdx.x;
    const int lane  = tid & 31;
    const int wave  = tid >> 5;
    const int col16 = lane & 15;
    const int half  = lane >> 4;
    const int q0    = blockIdx.x * 64;
    const int h     = blockIdx.y;
    const int b     = blockIdx.z;
    const int kv    = h / (NHEAD / NKV);
    const int window = *winp;
    const float scale = 0.125f;                        // 1/sqrt(64)

    const unsigned short* kh = kb + ((size_t)b * NKV + kv) * SEQ * HDIM;
    const unsigned short* vh = vb + ((size_t)b * NKV + kv) * SEQ * HDIM;

    // ---- per-wave Q fragments (A layout), kept for the whole kernel ----
    Frag16 aq[2];
    {
        const unsigned short* qp =
            qb + (((size_t)b * NHEAD + h) * SEQ + q0 + wave * 16 + col16) * HDIM;
        #pragma unroll
        for (int df = 0; df < 2; ++df)
            #pragma unroll
            for (int p = 0; p < 8; ++p)
                aq[df].w32[p] =
                    *(const unsigned int*)(qp + df * 32 + a_kbase(p, half));
    }

    v8f oacc[4];
    const v8f vzero = {0.f, 0.f, 0.f, 0.f, 0.f, 0.f, 0.f, 0.f};
    #pragma unroll
    for (int dt = 0; dt < 4; ++dt) oacc[dt] = vzero;
    float m_i[8], l_i[8];
    #pragma unroll
    for (int v = 0; v < 8; ++v) { m_i[v] = -1e30f; l_i[v] = 0.f; }

    const int klo = (q0 - window) > 0 ? (q0 - window) : 0;
    const int stg_key = tid >> 3, stg_ch = (tid & 7) * 8;  // 16B chunk coords

    for (int kt0 = klo & ~63; kt0 <= q0 + 63; kt0 += 64) {
        __syncthreads();                       // previous tile fully consumed
        // ---- async-stage K and V tiles: 64 keys x 128B each ----
        #pragma unroll
        for (int it = 0; it < 4; ++it) {
            int key = it * 16 + stg_key;
            async_copy_b128(kh + (size_t)(kt0 + key) * HDIM + stg_ch,
                            &sK[key * AT_KS + stg_ch]);
            async_copy_b128(vh + (size_t)(kt0 + key) * HDIM + stg_ch,
                            &sV[key * AT_KS + stg_ch]);
        }
        async_wait0();
        __syncthreads();                       // K,V visible to all waves

        // ---- S = Q K^T for 4 key sub-tiles of 16 (b128 LDS frag reads) ----
        float S[4][8];
        #pragma unroll
        for (int kt = 0; kt < 4; ++kt) {
            Frag16 bk[2];
            #pragma unroll
            for (int df = 0; df < 2; ++df) {
                const unsigned short* fp =
                    &sK[(kt * 16 + col16) * AT_KS + df * 32 + 8 * half];
                bk[df].q[0] = *(const uint4*)(fp);
                bk[df].q[1] = *(const uint4*)(fp + 16);
            }
            v8f s = vzero;
            s = __builtin_amdgcn_wmma_f32_16x16x32_bf16(
                    false, aq[0].v, false, bk[0].v, (short)0, s, false, false);
            s = __builtin_amdgcn_wmma_f32_16x16x32_bf16(
                    false, aq[1].v, false, bk[1].v, (short)0, s, false, false);
            #pragma unroll
            for (int v = 0; v < 8; ++v) {
                int qr   = q0 + wave * 16 + v + 8 * half;
                int diff = qr - (kt0 + kt * 16 + col16);
                S[kt][v] = (diff >= 0 && diff <= window) ? s[v] * scale
                                                         : -INFINITY;
            }
        }

        // ---- online softmax (16-lane butterflies keep halves separate) ----
        float alpha[8], rs[8];
        #pragma unroll
        for (int v = 0; v < 8; ++v) {
            float mv = m_i[v];
            #pragma unroll
            for (int kt = 0; kt < 4; ++kt) mv = fmaxf(mv, S[kt][v]);
            #pragma unroll
            for (int off = 1; off < 16; off <<= 1)
                mv = fmaxf(mv, __shfl_xor(mv, off, 32));
            alpha[v] = __expf(m_i[v] - mv);
            m_i[v]   = mv;
            rs[v]    = 0.f;
        }
        unsigned short* pw = sP[wave];
        #pragma unroll
        for (int kt = 0; kt < 4; ++kt)
            #pragma unroll
            for (int v = 0; v < 8; ++v) {
                float p = __expf(S[kt][v] - m_i[v]);
                rs[v] += p;
                pw[(v + 8 * half) * AT_PS + kt * 16 + col16] = bf16_bits(p);
            }
        #pragma unroll
        for (int v = 0; v < 8; ++v) {
            float r = rs[v];
            #pragma unroll
            for (int off = 1; off < 16; off <<= 1)
                r += __shfl_xor(r, off, 32);
            l_i[v] = l_i[v] * alpha[v] + r;
        }
        #pragma unroll
        for (int dt = 0; dt < 4; ++dt)
            #pragma unroll
            for (int v = 0; v < 8; ++v) oacc[dt][v] *= alpha[v];

        __syncthreads();                       // P visible (S reads done)

        // ---- O += P V : A frags from sP, B frags gathered from sV ----
        Frag16 ap[2];
        #pragma unroll
        for (int kf = 0; kf < 2; ++kf)
            #pragma unroll
            for (int p = 0; p < 8; ++p)
                ap[kf].w32[p] = *(const unsigned int*)
                    (&pw[col16 * AT_PS + kf * 32 + a_kbase(p, half)]);
        #pragma unroll
        for (int dt = 0; dt < 4; ++dt) {
            #pragma unroll
            for (int kf = 0; kf < 2; ++kf) {
                Frag16 bv;
                #pragma unroll
                for (int i = 0; i < 16; ++i)
                    bv.u[i] = sV[(kf * 32 + 8 * half + i + (i & 8)) * AT_KS +
                                 dt * 16 + col16];
                oacc[dt] = __builtin_amdgcn_wmma_f32_16x16x32_bf16(
                    false, ap[kf].v, false, bv.v, (short)0, oacc[dt],
                    false, false);
            }
        }
    }

    // ---- normalize and write y[b][q][h*64+d] (fp32, GEMM-ready) ----
    float* yp = y + ((size_t)b * SEQ + q0 + wave * 16 + 8 * half) * CDIM
                  + h * HDIM + col16;
    #pragma unroll
    for (int v = 0; v < 8; ++v) {
        float inv = 1.0f / l_i[v];
        yp[0]  = oacc[0][v] * inv;
        yp[16] = oacc[1][v] * inv;
        yp[32] = oacc[2][v] * inv;
        yp[48] = oacc[3][v] * inv;
        yp += CDIM;
    }
}

// ---------------------------------------------------------------------------
// Host launcher
// ---------------------------------------------------------------------------
extern "C" void kernel_launch(void* const* d_in, const int* in_sizes, int n_in,
                              void* d_out, int out_size, void* d_ws,
                              size_t ws_size, hipStream_t stream) {
    const float* x      = (const float*)d_in[0];
    const float* ve     = (const float*)d_in[1];
    const float* cosb   = (const float*)d_in[2];
    const float* sinb   = (const float*)d_in[3];
    const float* wq     = (const float*)d_in[4];
    const float* wk     = (const float*)d_in[5];
    const float* wv     = (const float*)d_in[6];
    const float* wo     = (const float*)d_in[7];
    const float* w_gate = (const float*)d_in[8];
    const int*   winp   = (const int*)d_in[9];
    float* out = (float*)d_out;

    // workspace partition (qkv scratch is reused as the attention output y)
    char* ws = (char*)d_ws;
    float*          qkv = (float*)ws;                       // 8192*1536 f32
    size_t off = (size_t)NTOK * QKVW * sizeof(float);
    unsigned short* qb  = (unsigned short*)(ws + off);      // 8192*1024 bf16
    off += (size_t)NTOK * CDIM * sizeof(unsigned short);
    unsigned short* kb  = (unsigned short*)(ws + off);      // 8192*256 bf16
    off += (size_t)NTOK * NKV * HDIM * sizeof(unsigned short);
    unsigned short* vb  = (unsigned short*)(ws + off);      // 8192*256 bf16
    float* y = qkv;                                         // alias (dead qkv)

    // 1) QKV projections into a single 1536-wide scratch
    gemm_bf16_wmma<<<dim3(CDIM / GM_BN, NTOK / GM_BM), 256, 0, stream>>>(
        x, wq, qkv, CDIM, QKVW, 0);
    gemm_bf16_wmma<<<dim3((NKV * HDIM) / GM_BN, NTOK / GM_BM), 256, 0, stream>>>(
        x, wk, qkv, CDIM, QKVW, CDIM);
    gemm_bf16_wmma<<<dim3((NKV * HDIM) / GM_BN, NTOK / GM_BM), 256, 0, stream>>>(
        x, wv, qkv, CDIM, QKVW, CDIM + NKV * HDIM);

    // 2) rotary / rms / gated-ve epilogue -> bf16 q,k,v
    {
        int jobs = NTOK * (NHEAD + 2 * NKV);
        qkv_epilogue<<<(jobs + 255) / 256, 256, 0, stream>>>(
            qkv, x, ve, cosb, sinb, w_gate, qb, kb, vb);
    }

    // 3) sliding-window flash attention
    attn_window_wmma<<<dim3(SEQ / 64, NHEAD, BATCH), 128, 0, stream>>>(
        qb, kb, vb, y, winp);

    // 4) output projection
    gemm_bf16_wmma<<<dim3(CDIM / GM_BN, NTOK / GM_BM), 256, 0, stream>>>(
        y, wo, out, CDIM, CDIM, 0);
}
